// MultiHeadMemoryBank_40931038330964
// MI455X (gfx1250) — matrix operand order repaired and checked by reading, not compiled, not run on hardware
//
#include <hip/hip_runtime.h>
#include <hip/hip_bf16.h>
#include <math.h>

// ---------------------------------------------------------------------------
// MultiHeadMemoryBank on MI455X (gfx1250, wave32).
// B=32, N=2048, D=256, H=8, BD=64, TOPK=16.
// Strategy: normalize memory once -> f16 copy (halves sim-pass bytes);
// fused read+write similarity GEMM via v_wmma_f32_16x16x32_f16 (A = 16 rows:
// 8 read keys + 8 write keys, both l2-normalized and pre-scaled by beta);
// all streaming passes coalesced VALU at the HBM roofline with prefetch.
// ---------------------------------------------------------------------------

typedef __attribute__((ext_vector_type(16))) _Float16 v16h;
typedef __attribute__((ext_vector_type(8)))  _Float16 v8h;
typedef __attribute__((ext_vector_type(8)))  float    v8f;

#define B_  32
#define N_  2048
#define D_  256
#define H_  8
#define BD_ 64
#define TOPK_ 16

// ---- block reductions (blockDim.x == 256 -> 8 wave32) ---------------------
__device__ __forceinline__ float block_sum(float v, float* sred) {
  #pragma unroll
  for (int o = 16; o > 0; o >>= 1) v += __shfl_xor(v, o, 32);
  int lane = threadIdx.x & 31, wv = threadIdx.x >> 5;
  if (lane == 0) sred[wv] = v;
  __syncthreads();
  float t = sred[0] + sred[1] + sred[2] + sred[3] +
            sred[4] + sred[5] + sred[6] + sred[7];
  __syncthreads();
  return t;
}

__device__ __forceinline__ float block_max(float v, float* sred) {
  #pragma unroll
  for (int o = 16; o > 0; o >>= 1) v = fmaxf(v, __shfl_xor(v, o, 32));
  int lane = threadIdx.x & 31, wv = threadIdx.x >> 5;
  if (lane == 0) sred[wv] = v;
  __syncthreads();
  float t = fmaxf(fmaxf(fmaxf(sred[0], sred[1]), fmaxf(sred[2], sred[3])),
                  fmaxf(fmaxf(sred[4], sred[5]), fmaxf(sred[6], sred[7])));
  __syncthreads();
  return t;
}

// ---- K1: l2-normalize memory rows -> f16; zero scratch accumulators -------
__global__ void k_norm_mem(const float* __restrict__ mem,
                           _Float16* __restrict__ m16,
                           float* __restrict__ mean_we,
                           float* __restrict__ agemax) {
  __shared__ float sred[8];
  int row = blockIdx.x;                 // b*N + n
  int d = threadIdx.x;
  float v = mem[(size_t)row * D_ + d];
  float ss = block_sum(v * v, sred);
  float inv = 1.f / fmaxf(sqrtf(ss), 1e-12f);
  m16[(size_t)row * D_ + d] = (_Float16)(v * inv);
  if (d == 0) {
    mean_we[row] = 0.f;
    if (row == 0) *agemax = 0.f;
  }
}

// ---- K2: build 16-row f16 A-matrix: rows 0-7 read keys, 8-15 write keys ---
__global__ void k_prep_keys(const float* __restrict__ rk,
                            const float* __restrict__ wk,
                            const float* __restrict__ beta,
                            _Float16* __restrict__ key16) {
  __shared__ float sred[8];
  int b = blockIdx.x, d = threadIdx.x;
  for (int row = 0; row < 16; ++row) {
    int h = row & 7;
    const float* src = (row < 8 ? rk : wk) + ((size_t)(b * H_ + h)) * D_;
    float v = src[d];
    float ss = block_sum(v * v, sred);
    float scale = beta[b * H_ + h] / fmaxf(sqrtf(ss), 1e-12f);
    key16[((size_t)b * 16 + row) * D_ + d] = (_Float16)(v * scale);
  }
}

// ---- K3: read-policy gate = softmax(read_keys @ Wg + bg) ------------------
__global__ void k_gate(const float* __restrict__ rk,
                       const float* __restrict__ Wg,
                       const float* __restrict__ bg,
                       float* __restrict__ gate) {
  __shared__ float sred[8];
  int bh = blockIdx.x, d = threadIdx.x;
  float k = rk[(size_t)bh * D_ + d];
  float g0 = block_sum(k * Wg[d * 3 + 0], sred);
  float g1 = block_sum(k * Wg[d * 3 + 1], sred);
  float g2 = block_sum(k * Wg[d * 3 + 2], sred);
  if (d == 0) {
    g0 += bg[0]; g1 += bg[1]; g2 += bg[2];
    float m = fmaxf(g0, fmaxf(g1, g2));
    float e0 = expf(g0 - m), e1 = expf(g1 - m), e2 = expf(g2 - m);
    float inv = 1.f / (e0 + e1 + e2);
    gate[bh * 3 + 0] = e0 * inv;
    gate[bh * 3 + 1] = e1 * inv;
    gate[bh * 3 + 2] = e2 * inv;
  }
}

// ---- K4: global max of (age + 1) (scalar, positive -> bitwise atomicMax) --
__global__ void k_agemax(const float* __restrict__ age,
                         float* __restrict__ agemax) {
  __shared__ float sred[8];
  int b = blockIdx.x, t = threadIdx.x;
  float m = 0.f;
  for (int n = t; n < N_; n += 256) m = fmaxf(m, age[(size_t)b * N_ + n] + 1.f);
  m = block_max(m, sred);
  if (t == 0) atomicMax((unsigned int*)agemax, __float_as_uint(m));
}

// ---- K5: fused read+write similarity via WMMA -----------------------------
// One wave per 16-column tile of one batch. A = key16[b] (16 x 256 f16),
// B^T rows = m16[b] (2048 x 256 f16). C lanes 0-15 -> sim_r, 16-31 -> sim_w.
__global__ void k_sim_wmma(const _Float16* __restrict__ key16,
                           const _Float16* __restrict__ m16,
                           float* __restrict__ sim_r,
                           float* __restrict__ sim_w) {
  int b = blockIdx.x;
  int wave = threadIdx.x >> 5, lane = threadIdx.x & 31;
  int ntile = blockIdx.y * 8 + wave;
  int col = ntile * 16 + (lane & 15);

  const _Float16* arow = key16 + ((size_t)b * 16 + (lane & 15)) * D_;
  const _Float16* brow = m16 + ((size_t)b * N_ + col) * D_;
  // ISA 16-bit A 16x32 layout: lanes<16 -> K {0..7,16..23}; lanes>=16 -> +8.
  int ka = (lane < 16) ? 0 : 8;
  // ISA 16-bit B 32x16 layout: lanes<16 -> K {0..7,16..23}; lanes>=16 -> +8.
  int kb = (lane < 16) ? 0 : 8;

  v8f c = {};
  #pragma unroll
  for (int kk = 0; kk < 8; ++kk) {
    int k0 = kk * 32;
    v8h alo = *(const v8h*)(arow + k0 + ka);
    v8h ahi = *(const v8h*)(arow + k0 + ka + 16);
    v8h blo = *(const v8h*)(brow + k0 + kb);
    v8h bhi = *(const v8h*)(brow + k0 + kb + 16);
    v16h a, bf;
    #pragma unroll
    for (int i = 0; i < 8; ++i) { a[i] = alo[i]; a[8 + i] = ahi[i]; }
    #pragma unroll
    for (int i = 0; i < 8; ++i) { bf[i] = blo[i]; bf[8 + i] = bhi[i]; }
    c = __builtin_amdgcn_wmma_f32_16x16x32_f16(
        /*neg_a=*/false, a, /*neg_b=*/false, bf,
        /*c_mod=*/(short)0, c, /*reuse_a=*/false, /*reuse_b=*/false);
  }
  // C layout: lanes 0-15 VGPR r -> M=r (read heads); lanes 16-31 -> M=8+r.
  float* dst = (lane < 16) ? sim_r : sim_w;
  #pragma unroll
  for (int r = 0; r < H_; ++r)
    dst[((size_t)b * H_ + r) * N_ + col] = c[r];
}

// ---- shared top-16 sparse softmax over one 2048-row in LDS ----------------
__device__ __forceinline__ void topk_softmax_lds(float* s_sim, float* sv, int* si,
                                                 float* s_topv, int* s_topi,
                                                 float* s_scl) {
  int t = threadIdx.x;
  for (int it = 0; it < TOPK_; ++it) {
    float lv = -__builtin_inff(); int li = 0;
    #pragma unroll
    for (int j = 0; j < 8; ++j) {
      int n = t + j * 256;
      float v = s_sim[n];
      if (v > lv) { lv = v; li = n; }
    }
    sv[t] = lv; si[t] = li;
    __syncthreads();
    for (int s = 128; s > 0; s >>= 1) {
      if (t < s && sv[t + s] > sv[t]) { sv[t] = sv[t + s]; si[t] = si[t + s]; }
      __syncthreads();
    }
    if (t == 0) {
      s_topv[it] = sv[0]; s_topi[it] = si[0];
      s_sim[si[0]] = -__builtin_inff();
    }
    __syncthreads();
  }
  if (t == 0) {
    float m = s_topv[0], s = 0.f;
    for (int i = 0; i < TOPK_; ++i) s += expf(s_topv[i] - m);
    s_scl[0] = m; s_scl[1] = 1.f / s;
  }
  __syncthreads();
  // scatter sparse softmax weights back into s_sim (zeros elsewhere)
  #pragma unroll
  for (int j = 0; j < 8; ++j) s_sim[t + j * 256] = 0.f;
  __syncthreads();
  if (t < TOPK_) s_sim[s_topi[t]] = expf(s_topv[t] - s_scl[0]) * s_scl[1];
  __syncthreads();
}

// ---- K6: read weights: top-k softmax + gate mix with uniform/random -------
__global__ void k_read_weights(const float* __restrict__ sim_r,
                               const float* __restrict__ noise,
                               const float* __restrict__ gate,
                               float* __restrict__ final_w) {
  __shared__ float s_sim[N_];
  __shared__ float sv[256]; __shared__ int si[256];
  __shared__ float s_topv[TOPK_]; __shared__ int s_topi[TOPK_];
  __shared__ float s_scl[2];
  __shared__ float sred[8];
  int bh = blockIdx.x, t = threadIdx.x;
  const float* srow = sim_r + (size_t)bh * N_;
  const float* nrow = noise + (size_t)bh * N_;
  float rn[8];
  #pragma unroll
  for (int j = 0; j < 8; ++j) {
    s_sim[t + j * 256] = srow[t + j * 256];
    rn[j] = nrow[t + j * 256];
  }
  __syncthreads();
  topk_softmax_lds(s_sim, sv, si, s_topv, s_topi, s_scl);
  // softmax over noise
  float lm = -__builtin_inff();
  #pragma unroll
  for (int j = 0; j < 8; ++j) lm = fmaxf(lm, rn[j]);
  float nm = block_max(lm, sred);
  float ls = 0.f;
  #pragma unroll
  for (int j = 0; j < 8; ++j) ls += expf(rn[j] - nm);
  float inv_ns = 1.f / block_sum(ls, sred);
  float g0 = gate[bh * 3 + 0], g1 = gate[bh * 3 + 1], g2 = gate[bh * 3 + 2];
  float u = g1 * (1.f / (float)N_);
  float* out = final_w + (size_t)bh * N_;
  #pragma unroll
  for (int j = 0; j < 8; ++j) {
    int n = t + j * 256;
    out[n] = g0 * s_sim[n] + u + g2 * expf(rn[j] - nm) * inv_ns;
  }
}

// ---- K7: read_ph[b,h,d] = sum_n final_w[b,h,n] * memory[b,n,d] ------------
__global__ void k_read_ph(const float* __restrict__ mem,
                          const float* __restrict__ fw,
                          float* __restrict__ read_ph) {
  __shared__ float s_w[H_][32];
  int b = blockIdx.x, t = threadIdx.x;   // t == d
  float acc[H_] = {0.f, 0.f, 0.f, 0.f, 0.f, 0.f, 0.f, 0.f};
  for (int n0 = 0; n0 < N_; n0 += 32) {
    { int h = t >> 5, nn = t & 31;
      s_w[h][nn] = fw[((size_t)b * H_ + h) * N_ + n0 + nn]; }
    __syncthreads();
    const float* mrow = mem + ((size_t)b * N_ + n0) * D_ + t;
    __builtin_prefetch(mrow + 32 * D_, 0, 1);   // global_prefetch_b8
    #pragma unroll 4
    for (int nn = 0; nn < 32; ++nn) {
      float m = mrow[(size_t)nn * D_];
      #pragma unroll
      for (int h = 0; h < H_; ++h) acc[h] = fmaf(s_w[h][nn], m, acc[h]);
    }
    __syncthreads();
  }
  #pragma unroll
  for (int h = 0; h < H_; ++h)
    read_ph[((size_t)b * H_ + h) * D_ + t] = acc[h];
}

// ---- K8: head merge (rc = read_ph @ Wm + bm) + LayerNorm ------------------
__global__ void k_merge_ln(const float* __restrict__ read_ph,
                           const float* __restrict__ Wm,
                           const float* __restrict__ bm,
                           const float* __restrict__ gamma,
                           const float* __restrict__ lbeta,
                           float* __restrict__ out) {
  __shared__ float s_r[H_ * D_];
  __shared__ float sred[8];
  int b = blockIdx.x, d = threadIdx.x;
  #pragma unroll
  for (int j = 0; j < 8; ++j)
    s_r[d + j * 256] = read_ph[(size_t)b * (H_ * D_) + d + j * 256];
  __syncthreads();
  float acc = bm[d];
  for (int k = 0; k < H_ * D_; ++k)
    acc = fmaf(s_r[k], Wm[(size_t)k * D_ + d], acc);
  float mu = block_sum(acc, sred) * (1.f / (float)D_);
  float dv = acc - mu;
  float var = block_sum(dv * dv, sred) * (1.f / (float)D_);
  out[(size_t)b * D_ + d] = dv * rsqrtf(var + 1e-5f) * gamma[d] + lbeta[d];
}

// ---- K9: compressed = gelu(write_vals @ W1 + b1) @ W2 + b2 ----------------
__global__ void k_write_mlp(const float* __restrict__ wv,
                            const float* __restrict__ W1,
                            const float* __restrict__ b1,
                            const float* __restrict__ W2,
                            const float* __restrict__ b2,
                            float* __restrict__ comp) {
  __shared__ float s_x[D_];
  __shared__ float s_h[BD_];
  int bh = blockIdx.x, t = threadIdx.x;
  s_x[t] = wv[(size_t)bh * D_ + t];
  __syncthreads();
  if (t < BD_) {
    float a = b1[t];
    for (int k = 0; k < D_; ++k) a = fmaf(s_x[k], W1[k * BD_ + t], a);
    s_h[t] = 0.5f * a * (1.f + erff(a * 0.70710678118654752f));  // exact GELU
  }
  __syncthreads();
  float a = b2[t];
  #pragma unroll 8
  for (int j = 0; j < BD_; ++j) a = fmaf(s_h[j], W2[j * D_ + t], a);
  comp[(size_t)bh * D_ + t] = a;
}

// ---- K10: write weights: (sim_w + age_bias) top-k softmax; mean erase -----
__global__ void k_write_weights(const float* __restrict__ sim_w,
                                const float* __restrict__ age,
                                const float* __restrict__ agemax,
                                const float* __restrict__ erase,
                                float* __restrict__ w_w_out,
                                float* __restrict__ mean_we) {
  __shared__ float s_sim[N_];
  __shared__ float sv[256]; __shared__ int si[256];
  __shared__ float s_topv[TOPK_]; __shared__ int s_topi[TOPK_];
  __shared__ float s_scl[2];
  int bh = blockIdx.x, t = threadIdx.x;
  int b = bh >> 3;
  float inv_g = 1.f / (*agemax + 1e-8f);
  const float* srow = sim_w + (size_t)bh * N_;
  const float* arow = age + (size_t)b * N_;
  #pragma unroll
  for (int j = 0; j < 8; ++j) {
    int n = t + j * 256;
    s_sim[n] = srow[n] + (arow[n] + 1.f) * inv_g;
  }
  __syncthreads();
  topk_softmax_lds(s_sim, sv, si, s_topv, s_topi, s_scl);
  float er = erase[bh] * (1.f / (float)H_);
  float* out = w_w_out + (size_t)bh * N_;
  #pragma unroll
  for (int j = 0; j < 8; ++j) {
    int n = t + j * 256;
    float w = s_sim[n];
    out[n] = w;
    if (w != 0.f) atomicAdd(&mean_we[(size_t)b * N_ + n], w * er);
  }
}

// ---- K11: new_mem = l2norm(mem*(1-mean_we) + add_term + 1e-8)*sigmoid(dg) -
__global__ void k_mem_update(const float* __restrict__ mem,
                             const float* __restrict__ mean_we,
                             const float* __restrict__ w_w,
                             const float* __restrict__ add_gate,
                             const float* __restrict__ comp,
                             const float* __restrict__ decay,
                             float* __restrict__ new_mem) {
  __shared__ float s_aw[H_];
  __shared__ float sred[8];
  int row = blockIdx.x;                 // b*N + n
  int b = row >> 11, n = row & (N_ - 1);
  int d = threadIdx.x;
  if (d < H_)
    s_aw[d] = w_w[((size_t)b * H_ + d) * N_ + n] * add_gate[b * H_ + d] *
              (1.f / (float)H_);
  __syncthreads();
  float we = mean_we[row];
  float v = mem[(size_t)row * D_ + d] * (1.f - we);
  #pragma unroll
  for (int h = 0; h < H_; ++h)
    v = fmaf(s_aw[h], comp[((size_t)b * H_ + h) * D_ + d], v);
  v += 1e-8f;
  float ss = block_sum(v * v, sred);
  float scale = (1.f / fmaxf(sqrtf(ss), 1e-12f)) *
                (1.f / (1.f + expf(-decay[n])));
  new_mem[(size_t)row * D_ + d] = v * scale;
}

// ---------------------------------------------------------------------------
extern "C" void kernel_launch(void* const* d_in, const int* in_sizes, int n_in,
                              void* d_out, int out_size, void* d_ws, size_t ws_size,
                              hipStream_t stream) {
  (void)in_sizes; (void)n_in; (void)out_size; (void)ws_size;
  const float* mem   = (const float*)d_in[0];
  const float* rk    = (const float*)d_in[1];
  const float* wk    = (const float*)d_in[2];
  const float* wv    = (const float*)d_in[3];
  const float* erase = (const float*)d_in[4];
  const float* addg  = (const float*)d_in[5];
  const float* beta  = (const float*)d_in[6];
  const float* age   = (const float*)d_in[7];
  const float* noise = (const float*)d_in[8];
  const float* Wg    = (const float*)d_in[9];
  const float* bg    = (const float*)d_in[10];
  const float* Wm    = (const float*)d_in[11];
  const float* bm    = (const float*)d_in[12];
  const float* lng   = (const float*)d_in[13];
  const float* lnb   = (const float*)d_in[14];
  const float* W1    = (const float*)d_in[15];
  const float* b1    = (const float*)d_in[16];
  const float* W2    = (const float*)d_in[17];
  const float* b2    = (const float*)d_in[18];
  const float* decay = (const float*)d_in[19];

  float* out = (float*)d_out;
  float* out_rc  = out;                                       // [B,D]
  float* out_mem = out + (size_t)B_ * D_;                     // [B,N,D]
  float* out_fw  = out_mem + (size_t)B_ * N_ * D_;            // [B,H,N]
  float* out_ww  = out_fw + (size_t)B_ * H_ * N_;             // [B,H,N]

  char* p = (char*)d_ws;
  auto take = [&](size_t bytes) {
    void* r = (void*)p;
    p += (bytes + 255) & ~(size_t)255;
    return r;
  };
  _Float16* m16   = (_Float16*)take((size_t)B_ * N_ * D_ * 2);   // 33.5 MB
  _Float16* key16 = (_Float16*)take((size_t)B_ * 16 * D_ * 2);
  float* sim_r    = (float*)take((size_t)B_ * H_ * N_ * 4);
  float* sim_w    = (float*)take((size_t)B_ * H_ * N_ * 4);
  float* gate     = (float*)take((size_t)B_ * H_ * 3 * 4);
  float* read_ph  = (float*)take((size_t)B_ * H_ * D_ * 4);
  float* comp     = (float*)take((size_t)B_ * H_ * D_ * 4);
  float* mean_we  = (float*)take((size_t)B_ * N_ * 4);
  float* agemax   = (float*)take(256);

  k_norm_mem     <<<B_ * N_,       256, 0, stream>>>(mem, m16, mean_we, agemax);
  k_prep_keys    <<<B_,            256, 0, stream>>>(rk, wk, beta, key16);
  k_gate         <<<B_ * H_,       256, 0, stream>>>(rk, Wg, bg, gate);
  k_agemax       <<<B_,            256, 0, stream>>>(age, agemax);
  k_sim_wmma     <<<dim3(B_, 16),  256, 0, stream>>>(key16, m16, sim_r, sim_w);
  k_write_mlp    <<<B_ * H_,       256, 0, stream>>>(wv, W1, b1, W2, b2, comp);
  k_read_weights <<<B_ * H_,       256, 0, stream>>>(sim_r, noise, gate, out_fw);
  k_read_ph      <<<B_,            256, 0, stream>>>(mem, out_fw, read_ph);
  k_merge_ln     <<<B_,            256, 0, stream>>>(read_ph, Wm, bm, lng, lnb, out_rc);
  k_write_weights<<<B_ * H_,       256, 0, stream>>>(sim_w, age, agemax, erase, out_ww, mean_we);
  k_mem_update   <<<B_ * N_,       256, 0, stream>>>(mem, mean_we, out_ww, addg, comp, decay, out_mem);
}
